// DeepARModel_54339926229332
// MI455X (gfx1250) — matrix operand order repaired and checked by reading, not compiled, not run on hardware
//
#include <hip/hip_runtime.h>
#include <hip/hip_bf16.h>
#include <math.h>

// ---------------------------------------------------------------------------
// DeepAR forward on MI455X (gfx1250), wave32 + WMMA f16 (16x16x32, f32 accum).
//
// Fragment layouts (C/D as in ISA 7.12.2; A/B use this code's contiguous k order):
//  A-frag (16x32 f16): lane l row M=l%16; halves [0..15] at K=(l/16)*16..+15
//    contiguous  -> two b128 loads per k-step.
//  B-frag (32x16 f16): lane l col N=l%16, K=(l/16)*16..+15 contiguous
//    -> weights pre-transposed f16 Bw[n][k]; two contiguous b128 loads.
//  C/D-frag: VGPR r, lane l -> M = r + 8*(l/16), N = l%16.
//
// Whole 2-layer, 192-step LSTM scan in ONE persistent kernel (64 WGs x 256
// thr = 512 waves; each wave owns one 16x16 (batch x hidden) tile and its 4
// gate accumulators). Cell state stays in VGPRs across the entire scan.
// Grid barrier: monotonic atomic counter + agent-scope fences, 2 per step.
//
// vs previous round: explicit address_space(1) pointers in the hot loop so
// weight loads lower to global_load_b128 (they were flat_load_b128, which
// consumes DScnt + the LDS path and blocks clause grouping).
// ---------------------------------------------------------------------------

#define B_DIM   256
#define T_DIM   192
#define F_DIM   8
#define E0_DIM  32
#define E1_DIM  16
#define H_DIM   512
#define G_DIM   2048          // 4*H
#define XK      64            // padded input feature K (56 -> 64)
#define LDK1    576           // layer-1 fused K: 512 (Wr) + 64 (Wk, padded)
#define LDK2    1024          // layer-2 fused K: 512 (Wr) + 512 (Wk)
#define BT      (B_DIM * T_DIM)      // 49152
#define NUMWG_SCAN 64

typedef __attribute__((ext_vector_type(8)))  _Float16 v8h;
typedef __attribute__((ext_vector_type(16))) _Float16 v16h;
typedef __attribute__((ext_vector_type(8)))  float    v8f;

// global (address_space 1) pointer flavors for the hot loop
typedef const _Float16 __attribute__((address_space(1)))* gch_p;
typedef       _Float16 __attribute__((address_space(1)))* gh_p;
typedef const v8h      __attribute__((address_space(1)))* gv8h_p;

// ---- fragment loaders (global address space -> global_load_b128) ----------
__device__ __forceinline__ v16h load_a_frag(gch_p p) {
  // 16 contiguous halves
  v8h lo = *(gv8h_p)p;
  v8h hi = *(gv8h_p)(p + 8);
  return __builtin_shufflevector(lo, hi, 0,1,2,3,4,5,6,7,8,9,10,11,12,13,14,15);
}
__device__ __forceinline__ v16h load_b_frag(gch_p p) {
  // 16 contiguous halves
  v8h lo = *(gv8h_p)p;
  v8h hi = *(gv8h_p)(p + 8);
  return __builtin_shufflevector(lo, hi, 0,1,2,3,4,5,6,7,8,9,10,11,12,13,14,15);
}

// ---- grid-wide barrier (persistent kernel, all WGs resident) ---------------
__device__ __forceinline__ void grid_barrier(unsigned* cnt, unsigned& target) {
  __builtin_amdgcn_fence(__ATOMIC_RELEASE, "agent");   // publish my h writes
  __syncthreads();
  if (threadIdx.x == 0) {
    __hip_atomic_fetch_add(cnt, 1u, __ATOMIC_RELAXED, __HIP_MEMORY_SCOPE_AGENT);
    const unsigned tgt = target;
    while (__hip_atomic_load(cnt, __ATOMIC_RELAXED, __HIP_MEMORY_SCOPE_AGENT) < tgt)
      __builtin_amdgcn_s_sleep(2);
  }
  __syncthreads();
  __builtin_amdgcn_fence(__ATOMIC_ACQUIRE, "agent");   // invalidate stale lines
  target += NUMWG_SCAN;
}

// ---- KS k-steps of the 4-gate GEMM from one A source -----------------------
// Rolled (unroll 2): issue A load + 4 B loads, then 4 WMMAs; partial
// s_wait_loadcnt lets wmma(q=0) overlap the later B loads.
template<int KS>
__device__ __forceinline__ void gemm_span(gch_p ap, gch_p (&bp)[4], v8f (&acc)[4])
{
#pragma unroll 2
  for (int kk = 0; kk < KS; ++kk) {
    const v16h A = load_a_frag(ap);
    ap += 32;
    v16h Bf[4];
#pragma unroll
    for (int q = 0; q < 4; ++q) Bf[q] = load_b_frag(bp[q]);
#pragma unroll
    for (int q = 0; q < 4; ++q) {
      bp[q] += 32;
      acc[q] = __builtin_amdgcn_wmma_f32_16x16x32_f16(
          false, A, false, Bf[q], (short)0, acc[q], false, false);
    }
  }
}

// ---- gates (Keras order i,f,g,o), cell update, h write ---------------------
__device__ __forceinline__ void gates_update(v8f (&acc)[4], float* __restrict__ c,
                                             gh_p hs_out,
                                             const int t, const int wrow, const int col)
{
#pragma unroll
  for (int e = 0; e < 8; ++e) {
    const float ig = 1.f / (1.f + __expf(-acc[0][e]));
    const float fg = 1.f / (1.f + __expf(-acc[1][e]));
    const float gg = tanhf(acc[2][e]);
    const float og = 1.f / (1.f + __expf(-acc[3][e]));
    const float cn = fg * c[e] + ig * gg;
    c[e] = cn;
    const float h = og * tanhf(cn);
    hs_out[((size_t)(wrow + e) * (T_DIM + 1) + (t + 1)) * H_DIM + col] = (_Float16)h;
  }
}

// ---- persistent scan kernel: both LSTM layers, all 192 steps ---------------
__global__ __launch_bounds__(256) void lstm_scan_kernel(
    const _Float16* __restrict__ x16,
    const _Float16* __restrict__ bw1, const _Float16* __restrict__ bw2,
    const float* __restrict__ b1, const float* __restrict__ b2,
    _Float16* __restrict__ hs1, _Float16* __restrict__ hs2,
    unsigned* __restrict__ counter)
{
  const int lane = threadIdx.x & 31;
  const int wv   = threadIdx.x >> 5;
  const int w    = blockIdx.x * 8 + wv;   // 0..511
  const int mt   = w >> 5;                // batch tile 0..15
  const int hc   = w & 31;                // hidden-col tile 0..31
  const int ln   = lane & 15;
  const int lh   = lane >> 4;

  // explicit global-AS views of the hot-loop buffers
  gch_p bw1g = (gch_p)bw1;
  gch_p bw2g = (gch_p)bw2;
  gch_p x16g = (gch_p)x16;
  gch_p h1r  = (gch_p)hs1;
  gch_p h2r  = (gch_p)hs2;
  gh_p  h1w  = (gh_p)hs1;
  gh_p  h2w  = (gh_p)hs2;

  // ---- loop-invariant hoists ----
  float bv1[4], bv2[4];
  gch_p bb1[4], bb2[4];
#pragma unroll
  for (int q = 0; q < 4; ++q) {
    const int ncol = q * H_DIM + hc * 16 + ln;
    bv1[q] = b1[ncol];
    bv2[q] = b2[ncol];
    bb1[q] = bw1g + (size_t)ncol * LDK1 + lh * 16;
    bb2[q] = bw2g + (size_t)ncol * LDK2 + lh * 16;
  }
  const int arow = mt * 16 + ln;          // A-fragment batch row for this lane
  const int wrow = mt * 16 + lh * 8;      // C-fragment batch row base
  const int col  = hc * 16 + ln;          // C-fragment hidden column
  gch_p a_h1 = h1r + (size_t)arow * (T_DIM + 1) * H_DIM + lh * 16;
  gch_p a_h2 = h2r + (size_t)arow * (T_DIM + 1) * H_DIM + lh * 16;
  gch_p a_x  = x16g + (size_t)arow * T_DIM * XK + lh * 16;

  float c1[8], c2[8];
#pragma unroll
  for (int e = 0; e < 8; ++e) { c1[e] = 0.f; c2[e] = 0.f; }

  unsigned target = NUMWG_SCAN;

#pragma unroll 1
  for (int t = 0; t < T_DIM; ++t) {
    // ---- layer 1: A = [ h1[t-1] | x16[t] ],  K = 512 + 64 ----
    {
      v8f acc[4];
      gch_p bp[4] = { bb1[0], bb1[1], bb1[2], bb1[3] };
#pragma unroll
      for (int q = 0; q < 4; ++q)
#pragma unroll
        for (int e = 0; e < 8; ++e) acc[q][e] = bv1[q];
      gemm_span<16>(a_h1 + (size_t)t * H_DIM, bp, acc);
      gemm_span<2>(a_x + (size_t)t * XK, bp, acc);
      gates_update(acc, c1, h1w, t, wrow, col);
    }
    grid_barrier(counter, target);

    // ---- layer 2: A = [ h2[t-1] | h1[t] ],  K = 512 + 512 ----
    {
      v8f acc[4];
      gch_p bp[4] = { bb2[0], bb2[1], bb2[2], bb2[3] };
#pragma unroll
      for (int q = 0; q < 4; ++q)
#pragma unroll
        for (int e = 0; e < 8; ++e) acc[q][e] = bv2[q];
      gemm_span<16>(a_h2 + (size_t)t * H_DIM, bp, acc);
      gemm_span<16>(a_h1 + (size_t)(t + 1) * H_DIM, bp, acc);
      gates_update(acc, c2, h2w, t, wrow, col);
    }
    grid_barrier(counter, target);
  }
}

// ---- prep: materialize f16 concat input [B][T][64] -------------------------
__global__ void prep_x16_kernel(const float* __restrict__ xc,
                                const int* __restrict__ c0,
                                const int* __restrict__ c1,
                                const float* __restrict__ e0,
                                const float* __restrict__ e1,
                                _Float16* __restrict__ x16)
{
  const int bt = blockIdx.x * 256 + threadIdx.x;   // < 49152
  const float* xrow = xc + (size_t)bt * F_DIM;
  const int i0 = c0[bt], i1 = c1[bt];
  _Float16* o = x16 + (size_t)bt * XK;
#pragma unroll
  for (int e = 0; e < F_DIM; ++e)  o[e]          = (_Float16)xrow[e];
#pragma unroll
  for (int e = 0; e < E0_DIM; ++e) o[F_DIM + e]  = (_Float16)e0[(size_t)i0 * E0_DIM + e];
#pragma unroll
  for (int e = 0; e < E1_DIM; ++e) o[F_DIM + E0_DIM + e] = (_Float16)e1[(size_t)i1 * E1_DIM + e];
#pragma unroll
  for (int e = F_DIM + E0_DIM + E1_DIM; e < XK; ++e) o[e] = (_Float16)0.f;
}

// ---- prep: fused+transposed f16 weights Bw[n][k] = [Wr ; Wk](k, n) ---------
__global__ void prep_bw_kernel(const float* __restrict__ Wk1, const float* __restrict__ Wr1,
                               const float* __restrict__ Wk2, const float* __restrict__ Wr2,
                               _Float16* __restrict__ bw1, _Float16* __restrict__ bw2)
{
  const int idx = blockIdx.x * 256 + threadIdx.x;
  const int N1 = G_DIM * LDK1;                    // 2048*576
  if (idx < N1) {
    const int n = idx / LDK1, k = idx % LDK1;
    float v = 0.f;
    if (k < H_DIM)                       v = Wr1[(size_t)k * G_DIM + n];
    else if (k < H_DIM + F_DIM + E0_DIM + E1_DIM)
                                         v = Wk1[(size_t)(k - H_DIM) * G_DIM + n];
    bw1[idx] = (_Float16)v;
  } else {
    const int j = idx - N1;                       // < 2048*1024
    const int n = j >> 10, k = j & 1023;
    const float v = (k < H_DIM) ? Wr2[(size_t)k * G_DIM + n]
                                : Wk2[(size_t)(k - H_DIM) * G_DIM + n];
    bw2[j] = (_Float16)v;
  }
}

// ---- init: zero barrier counter + t=-1 hidden-state slots ------------------
__global__ void init_kernel(_Float16* __restrict__ hs1, _Float16* __restrict__ hs2,
                            unsigned* __restrict__ counter)
{
  const int idx = blockIdx.x * 256 + threadIdx.x;  // < 131072 = 256*512
  if (idx == 0) *counter = 0u;
  const int b = idx >> 9, j = idx & 511;
  hs1[((size_t)b * (T_DIM + 1)) * H_DIM + j] = (_Float16)0.f;
  hs2[((size_t)b * (T_DIM + 1)) * H_DIM + j] = (_Float16)0.f;
}

// ---- heads: mu and softplus(sigma), one wave per (b,t) ---------------------
__global__ void heads_kernel(const _Float16* __restrict__ hs2,
                             const float* __restrict__ Wmu, const float* __restrict__ bmu,
                             const float* __restrict__ Wsig, const float* __restrict__ bsig,
                             float* __restrict__ out)
{
  const int wv = threadIdx.x >> 5, lane = threadIdx.x & 31;
  const int bt = blockIdx.x * 8 + wv;              // < 49152
  const int b = bt / T_DIM, t = bt % T_DIM;
  const _Float16* h = hs2 + ((size_t)b * (T_DIM + 1) + t + 1) * H_DIM;
  float smu = 0.f, ssg = 0.f;
#pragma unroll 4
  for (int i = lane; i < H_DIM; i += 32) {
    const float hv = (float)h[i];
    smu += hv * Wmu[i];
    ssg += hv * Wsig[i];
  }
#pragma unroll
  for (int off = 16; off; off >>= 1) {
    smu += __shfl_xor(smu, off);
    ssg += __shfl_xor(ssg, off);
  }
  if (lane == 0) {
    out[bt] = smu + bmu[0];
    const float z = ssg + bsig[0];
    out[BT + bt] = (z > 20.f) ? z : log1pf(__expf(z));
  }
}

// ---------------------------------------------------------------------------
extern "C" void kernel_launch(void* const* d_in, const int* in_sizes, int n_in,
                              void* d_out, int out_size, void* d_ws, size_t ws_size,
                              hipStream_t stream)
{
  (void)in_sizes; (void)n_in; (void)out_size;
  const float* x_cont = (const float*)d_in[0];
  const int*   cat0   = (const int*)  d_in[1];
  const int*   cat1   = (const int*)  d_in[2];
  const float* emb0   = (const float*)d_in[3];
  const float* emb1   = (const float*)d_in[4];
  const float* Wk1    = (const float*)d_in[5];
  const float* Wr1    = (const float*)d_in[6];
  const float* b1     = (const float*)d_in[7];
  const float* Wk2    = (const float*)d_in[8];
  const float* Wr2    = (const float*)d_in[9];
  const float* b2     = (const float*)d_in[10];
  const float* Wmu    = (const float*)d_in[11];
  const float* bmu    = (const float*)d_in[12];
  const float* Wsig   = (const float*)d_in[13];
  const float* bsig   = (const float*)d_in[14];
  float* out = (float*)d_out;

  // workspace carve-up (~114 MB total)
  char* ws = (char*)d_ws;
  size_t off = 0;
  unsigned* counter = (unsigned*)(ws + off);        off += 256;
  _Float16* x16 = (_Float16*)(ws + off);            off += (size_t)BT * XK * 2;            // 6 MB
  _Float16* bw1 = (_Float16*)(ws + off);            off += (size_t)G_DIM * LDK1 * 2;       // 2.25 MB
  _Float16* bw2 = (_Float16*)(ws + off);            off += (size_t)G_DIM * LDK2 * 2;       // 4 MB
  _Float16* hs1 = (_Float16*)(ws + off);            off += (size_t)B_DIM * (T_DIM + 1) * H_DIM * 2; // 50.6 MB
  _Float16* hs2 = (_Float16*)(ws + off);            off += (size_t)B_DIM * (T_DIM + 1) * H_DIM * 2; // 50.6 MB
  (void)ws_size;

  init_kernel<<<512, 256, 0, stream>>>(hs1, hs2, counter);
  prep_x16_kernel<<<BT / 256, 256, 0, stream>>>(x_cont, cat0, cat1, emb0, emb1, x16);
  prep_bw_kernel<<<(G_DIM * (LDK1 + LDK2)) / 256, 256, 0, stream>>>(Wk1, Wr1, Wk2, Wr2, bw1, bw2);
  lstm_scan_kernel<<<NUMWG_SCAN, 256, 0, stream>>>(x16, bw1, bw2, b1, b2, hs1, hs2, counter);
  heads_kernel<<<BT / 8, 256, 0, stream>>>(hs2, Wmu, bmu, Wsig, bsig, out);
}
